// SelfAttentionBlock_29076928594355
// MI455X (gfx1250) — matrix operand order repaired and checked by reading, not compiled, not run on hardware
//
#include <hip/hip_runtime.h>
#include <hip/hip_bf16.h>
#include <math.h>

#define B_  32
#define N_  512
#define D_  512
#define H_  8
#define HD_ 64
#define ED_ 2048
#define EPS_ 1e-3f

typedef __bf16 bf16;
typedef bf16  v16bf __attribute__((ext_vector_type(16)));
typedef bf16  v8bf  __attribute__((ext_vector_type(8)));
typedef float v8f   __attribute__((ext_vector_type(8)));
typedef unsigned int u32x4 __attribute__((ext_vector_type(4)));
typedef int i32x8 __attribute__((ext_vector_type(8)));
typedef int i32x4 __attribute__((ext_vector_type(4)));

// ---------------------------------------------------------------- WMMA helpers

static __device__ inline v8f wmma_bf16(v16bf a, v16bf b, v8f c) {
  // (neg_a, A, neg_b, B, c_mod, C, reuse_a, reuse_b)
  return __builtin_amdgcn_wmma_f32_16x16x32_bf16(false, a, false, b, (short)0, c,
                                                 false, false);
}

static __device__ inline v8f zero_v8f() {
  v8f z;
#pragma unroll
  for (int i = 0; i < 8; ++i) z[i] = 0.0f;
  return z;
}

// A-matrix 16x32 bf16 fragment (ISA 7.12.2): row = lane%16,
// elements 0..7 -> K = (lane/16)*8 + e ; elements 8..15 -> K = 16 + (lane/16)*8 + e.
static __device__ inline v16bf load_fragA(const bf16* __restrict__ base, int stride,
                                          int lane) {
  const bf16* p = base + (size_t)(lane & 15) * stride + ((lane >> 4) << 3);
  v8bf lo = *reinterpret_cast<const v8bf*>(p);
  v8bf hi = *reinterpret_cast<const v8bf*>(p + 16);
  v16bf r;
#pragma unroll
  for (int i = 0; i < 8; ++i) { r[i] = lo[i]; r[i + 8] = hi[i]; }
  return r;
}

// B-matrix 32x16 bf16 fragment from a K-contiguous (pre-transposed) layout:
// col = lane%16, elements 0..7 -> K = (lane/16)*8 + e ; elements 8..15 -> K = 16 + (lane/16)*8 + e.
static __device__ inline v16bf load_fragB(const bf16* __restrict__ base, int stride,
                                          int lane) {
  const bf16* p = base + (size_t)(lane & 15) * stride + ((lane >> 4) << 3);
  v8bf lo = *reinterpret_cast<const v8bf*>(p);
  v8bf hi = *reinterpret_cast<const v8bf*>(p + 16);
  v16bf r;
#pragma unroll
  for (int i = 0; i < 8; ++i) { r[i] = lo[i]; r[i + 8] = hi[i]; }
  return r;
}

// ---------------------------------------------------------------- TDM helper
// 2D tile load: rows x 32 bf16 elements, row stride = stride_elems (bf16 units),
// packed into LDS contiguously (row-major, 32 elems = 64 B per row).
// D# packing per CDNA5 ISA §8.3/8.4 (count=1, type=2 "image", data_size=1 (2B)).
// 6-arg builtin form (clang-23 / therock headers): (g0, g1, g2, g3, g4, cpol).
static __device__ inline void tdm_load_tile(unsigned lds_off, const bf16* gsrc,
                                            int rows, int stride_elems) {
  unsigned long long ga = (unsigned long long)(uintptr_t)gsrc;
  u32x4 g0;
  g0[0] = 1u;                                            // count=1, user mode
  g0[1] = lds_off;                                       // lds_addr (bytes)
  g0[2] = (unsigned)(ga & 0xffffffffu);                  // global_addr[31:0]
  g0[3] = (unsigned)((ga >> 32) & 0x01ffffffu) | (2u << 30);  // addr[56:32]|type=2
  i32x8 g1;
  g1[0] = (1 << 16);                                     // data_size=1 -> 2 bytes
  g1[1] = (stride_elems & 0xffff) << 16;                 // tensor_dim0[15:0]
  g1[2] = ((stride_elems >> 16) & 0xffff) | ((rows & 0xffff) << 16); // d0 hi|d1 lo
  g1[3] = ((rows >> 16) & 0xffff) | (32 << 16);          // d1 hi | tile_dim0=32
  g1[4] = rows & 0xffff;                                 // tile_dim1=rows, tile_dim2=0
  g1[5] = stride_elems;                                  // tensor_dim0_stride[31:0]
  g1[6] = 0;                                             // stride0 hi | stride1 lo
  g1[7] = 0;                                             // stride1 hi
  i32x4 z4;
  z4[0] = 0; z4[1] = 0; z4[2] = 0; z4[3] = 0;
  i32x8 z8;
#pragma unroll
  for (int i = 0; i < 8; ++i) z8[i] = 0;
  __builtin_amdgcn_tensor_load_to_lds(g0, g1, z4, z4, z8, 0);
}

// ---------------------------------------------------------------- reductions

static __device__ inline float wave_reduce_sum(float v) {
#pragma unroll
  for (int off = 16; off > 0; off >>= 1) v += __shfl_xor(v, off, 32);
  return v;
}

// ---------------------------------------------------------------- weight prep

__global__ __launch_bounds__(256) void convert_wt_kernel(
    const float* __restrict__ W, bf16* __restrict__ Wt, int K, int Nout) {
  int idx = blockIdx.x * 256 + threadIdx.x;
  if (idx >= K * Nout) return;
  int n = idx / K, k = idx - n * K;          // output is Wt[n][k] = W[k][n]
  Wt[(size_t)n * K + k] = (bf16)W[(size_t)k * Nout + n];
}

// vt[b][h][hd][key] = qkv[(b*N+key)*1536 + 2*512 + h*64 + hd]
__global__ __launch_bounds__(256) void vt_kernel(const bf16* __restrict__ qkv,
                                                 bf16* __restrict__ vt) {
  int idx = blockIdx.x * 256 + threadIdx.x;     // B*H*HD*N elements
  int key = idx & (N_ - 1);
  int hd  = (idx >> 9) & (HD_ - 1);
  int h   = (idx >> 15) & (H_ - 1);
  int b   = idx >> 18;
  vt[idx] = qkv[(size_t)(b * N_ + key) * (3 * D_) + 2 * D_ + h * HD_ + hd];
}

// ---------------------------------------------------------------- layernorms

__global__ __launch_bounds__(128) void ln512_bf16_kernel(
    const float* __restrict__ in, const float* __restrict__ g,
    const float* __restrict__ be, bf16* __restrict__ out) {
  __shared__ float red[4];
  const int row = blockIdx.x, tid = threadIdx.x;
  const float* p = in + (size_t)row * D_;
  float v[4];
#pragma unroll
  for (int i = 0; i < 4; ++i) v[i] = p[tid + i * 128];
  float s = wave_reduce_sum(v[0] + v[1] + v[2] + v[3]);
  if ((tid & 31) == 0) red[tid >> 5] = s;
  __syncthreads();
  const float mean = (red[0] + red[1] + red[2] + red[3]) * (1.0f / D_);
  __syncthreads();
  float sq = 0.f;
#pragma unroll
  for (int i = 0; i < 4; ++i) { v[i] -= mean; sq += v[i] * v[i]; }
  s = wave_reduce_sum(sq);
  if ((tid & 31) == 0) red[tid >> 5] = s;
  __syncthreads();
  const float inv = rsqrtf((red[0] + red[1] + red[2] + red[3]) * (1.0f / D_) + EPS_);
#pragma unroll
  for (int i = 0; i < 4; ++i) {
    int c = tid + i * 128;
    out[(size_t)row * D_ + c] = (bf16)(v[i] * inv * g[c] + be[c]);
  }
}

// attended = LN(oproj; g_post) + x  (stored f32)  ;  f = LN(attended; g_ffn) (bf16)
__global__ __launch_bounds__(128) void post_kernel(
    const float* __restrict__ oproj, const float* __restrict__ x,
    const float* __restrict__ gp, const float* __restrict__ bp,
    const float* __restrict__ gf, const float* __restrict__ bfv,
    float* __restrict__ attended, bf16* __restrict__ fout) {
  __shared__ float red[4];
  const int row = blockIdx.x, tid = threadIdx.x;
  float o[4], a[4];
#pragma unroll
  for (int i = 0; i < 4; ++i) o[i] = oproj[(size_t)row * D_ + tid + i * 128];
  float s = wave_reduce_sum(o[0] + o[1] + o[2] + o[3]);
  if ((tid & 31) == 0) red[tid >> 5] = s;
  __syncthreads();
  float mean = (red[0] + red[1] + red[2] + red[3]) * (1.0f / D_);
  __syncthreads();
  float sq = 0.f;
#pragma unroll
  for (int i = 0; i < 4; ++i) { o[i] -= mean; sq += o[i] * o[i]; }
  s = wave_reduce_sum(sq);
  if ((tid & 31) == 0) red[tid >> 5] = s;
  __syncthreads();
  float inv = rsqrtf((red[0] + red[1] + red[2] + red[3]) * (1.0f / D_) + EPS_);
  __syncthreads();
#pragma unroll
  for (int i = 0; i < 4; ++i) {
    int c = tid + i * 128;
    a[i] = o[i] * inv * gp[c] + bp[c] + x[(size_t)row * D_ + c];
    attended[(size_t)row * D_ + c] = a[i];
  }
  s = wave_reduce_sum(a[0] + a[1] + a[2] + a[3]);
  if ((tid & 31) == 0) red[tid >> 5] = s;
  __syncthreads();
  mean = (red[0] + red[1] + red[2] + red[3]) * (1.0f / D_);
  __syncthreads();
  sq = 0.f;
#pragma unroll
  for (int i = 0; i < 4; ++i) { a[i] -= mean; sq += a[i] * a[i]; }
  s = wave_reduce_sum(sq);
  if ((tid & 31) == 0) red[tid >> 5] = s;
  __syncthreads();
  inv = rsqrtf((red[0] + red[1] + red[2] + red[3]) * (1.0f / D_) + EPS_);
#pragma unroll
  for (int i = 0; i < 4; ++i) {
    int c = tid + i * 128;
    fout[(size_t)row * D_ + c] = (bf16)(a[i] * inv * gf[c] + bfv[c]);
  }
}

__global__ __launch_bounds__(256) void ln2048_bf16_kernel(
    const bf16* __restrict__ in, const float* __restrict__ g,
    const float* __restrict__ be, bf16* __restrict__ out) {
  __shared__ float red[8];
  const int row = blockIdx.x, tid = threadIdx.x;
  const bf16* p = in + (size_t)row * ED_;
  float v[8];
  float acc = 0.f;
#pragma unroll
  for (int i = 0; i < 8; ++i) { v[i] = (float)p[tid + i * 256]; acc += v[i]; }
  float s = wave_reduce_sum(acc);
  if ((tid & 31) == 0) red[tid >> 5] = s;
  __syncthreads();
  float mean = 0.f;
#pragma unroll
  for (int i = 0; i < 8; ++i) mean += red[i];
  mean *= (1.0f / ED_);
  __syncthreads();
  float sq = 0.f;
#pragma unroll
  for (int i = 0; i < 8; ++i) { v[i] -= mean; sq += v[i] * v[i]; }
  s = wave_reduce_sum(sq);
  if ((tid & 31) == 0) red[tid >> 5] = s;
  __syncthreads();
  float var = 0.f;
#pragma unroll
  for (int i = 0; i < 8; ++i) var += red[i];
  const float inv = rsqrtf(var * (1.0f / ED_) + EPS_);
#pragma unroll
  for (int i = 0; i < 8; ++i) {
    int c = tid + i * 256;
    out[(size_t)row * ED_ + c] = (bf16)(v[i] * inv * g[c] + be[c]);
  }
}

// ---------------------------------------------------------------- WMMA GEMM
// C[M,Nout] = A[M,K](bf16) x Wt[Nout,K](bf16 pre-transposed) + bias, epilogue.
// TDM double-buffered pipeline: wave 0 issues tensor_load_to_lds for the
// 128x32 A-tile and 64x32 B-tile; all 8 waves consume fragments from LDS.
// EPI 0: store bf16   EPI 1: store f32   EPI 2: GELU->bf16   EPI 3: +resid->f32
template <int EPI>
__global__ __launch_bounds__(256) void gemm_kernel(
    const bf16* __restrict__ A, const bf16* __restrict__ Wt,
    const float* __restrict__ bias, const float* __restrict__ resid,
    void* __restrict__ Cout, int M, int K, int Nout) {
  __shared__ bf16 ldsA[2][128 * 32];   // 8 KB x2
  __shared__ bf16 ldsB[2][64 * 32];    // 4 KB x2
  const int lane = threadIdx.x & 31;
  const int wave = threadIdx.x >> 5;
  const int rw = wave >> 1;                      // 0..3
  const int cw = wave & 1;                       // 0..1
  const int row0 = blockIdx.x * 128;
  const int col0 = blockIdx.y * 64;

  v8f acc[2][2];
  acc[0][0] = zero_v8f(); acc[0][1] = zero_v8f();
  acc[1][0] = zero_v8f(); acc[1][1] = zero_v8f();

  const bf16* a_base = A + (size_t)row0 * K;
  const bf16* b_base = Wt + (size_t)col0 * K;

  if (wave == 0) {
    tdm_load_tile((unsigned)(uintptr_t)&ldsA[0][0], a_base, 128, K);
    tdm_load_tile((unsigned)(uintptr_t)&ldsB[0][0], b_base, 64, K);
  }

  for (int kb = 0; kb < K; kb += 32) {
    const int cur = (kb >> 5) & 1;
    if (wave == 0) __builtin_amdgcn_s_wait_tensorcnt(0);
    __syncthreads();                               // buffer `cur` ready
    if (kb + 32 < K && wave == 0) {                // prefetch next tile pair
      tdm_load_tile((unsigned)(uintptr_t)&ldsA[1 - cur][0], a_base + kb + 32, 128, K);
      tdm_load_tile((unsigned)(uintptr_t)&ldsB[1 - cur][0], b_base + kb + 32, 64, K);
    }
    v16bf a0 = load_fragA(&ldsA[cur][(rw * 32) * 32], 32, lane);
    v16bf a1 = load_fragA(&ldsA[cur][(rw * 32 + 16) * 32], 32, lane);
    v16bf b0 = load_fragB(&ldsB[cur][(cw * 32) * 32], 32, lane);
    v16bf b1 = load_fragB(&ldsB[cur][(cw * 32 + 16) * 32], 32, lane);
    acc[0][0] = wmma_bf16(a0, b0, acc[0][0]);
    acc[0][1] = wmma_bf16(a0, b1, acc[0][1]);
    acc[1][0] = wmma_bf16(a1, b0, acc[1][0]);
    acc[1][1] = wmma_bf16(a1, b1, acc[1][1]);
    __syncthreads();                               // done reading `cur`
  }

#pragma unroll
  for (int i = 0; i < 2; ++i) {
#pragma unroll
    for (int j = 0; j < 2; ++j) {
#pragma unroll
      for (int r = 0; r < 8; ++r) {
        const int row = row0 + rw * 32 + i * 16 + r + ((lane >> 4) << 3);
        const int col = col0 + cw * 32 + j * 16 + (lane & 15);
        float v = acc[i][j][r] + bias[col];
        if (EPI == 2) v = 0.5f * v * (1.0f + erff(v * 0.70710678118f));
        const size_t oidx = (size_t)row * Nout + col;
        if (EPI == 3) v += resid[oidx];
        if (EPI == 0 || EPI == 2) ((bf16*)Cout)[oidx] = (bf16)v;
        else                      ((float*)Cout)[oidx] = v;
      }
    }
  }
}

// ---------------------------------------------------------------- attention
// grid (N/64, H, B), 128 threads = 4 waves; each wave owns 16 query rows.
__global__ __launch_bounds__(128) void attn_kernel(
    const bf16* __restrict__ qkv, const bf16* __restrict__ vt,
    const float* __restrict__ inter, const int* __restrict__ mask,
    bf16* __restrict__ o) {
  __shared__ float S[4][16][N_];                 // per-wave raw scores (128 KB)
  const int lane = threadIdx.x & 31;
  const int wave = threadIdx.x >> 5;
  const int bb = blockIdx.z;
  const int hh = blockIdx.y;
  const int q0 = blockIdx.x * 64 + wave * 16;

  // Q fragments (A layout), K = HD = 64 -> two k-steps
  const bf16* qbase = qkv + (size_t)(bb * N_ + q0) * (3 * D_) + hh * HD_;
  const v16bf qa0 = load_fragA(qbase, 3 * D_, lane);
  const v16bf qa1 = load_fragA(qbase + 32, 3 * D_, lane);

  // ---- scores: S = QK^T/8 + interaction + mask
  for (int t = 0; t < N_ / 16; ++t) {
    const int k0 = t * 16;
    const bf16* kbase =
        qkv + (size_t)(bb * N_ + k0) * (3 * D_) + D_ + hh * HD_;
    v16bf kb0 = load_fragB(kbase, 3 * D_, lane);
    v16bf kb1 = load_fragB(kbase + 32, 3 * D_, lane);
    v8f c = zero_v8f();
    c = wmma_bf16(qa0, kb0, c);
    c = wmma_bf16(qa1, kb1, c);
    const int nk = k0 + (lane & 15);
#pragma unroll
    for (int r = 0; r < 8; ++r) {
      const int m = r + ((lane >> 4) << 3);
      const int nq = q0 + m;
      const size_t pij = ((size_t)bb * N_ + nq) * N_ + nk;
      if (t + 1 < N_ / 16) __builtin_prefetch(&inter[(pij + 16) * H_ + hh], 0, 1);
      float s = c[r] * 0.125f + inter[pij * H_ + hh];
      s += mask[pij] ? 0.0f : -1e9f;
      S[wave][m][nk] = s;
    }
  }
  __syncthreads();

  // ---- row softmax stats: lane handles row = lane%16, half of the columns
  const int row = lane & 15;
  const int half = lane >> 4;
  const float* srow = &S[wave][row][0];
  float mx = -3.4e38f;
  for (int c = half * 256; c < half * 256 + 256; ++c) mx = fmaxf(mx, srow[c]);
  mx = fmaxf(mx, __shfl_xor(mx, 16, 32));
  float sm = 0.0f;
  for (int c = half * 256; c < half * 256 + 256; ++c) sm += __expf(srow[c] - mx);
  sm += __shfl_xor(sm, 16, 32);
  const float inv = 1.0f / sm;

  // ---- O = softmax(S) @ V ; V pre-transposed to [b][h][hd][key]
  for (int nt = 0; nt < HD_ / 16; ++nt) {
    v8f acc = zero_v8f();
    const bf16* vbase =
        vt + ((size_t)((bb * H_ + hh) * HD_) + nt * 16) * N_;
    for (int kb = 0; kb < N_; kb += 32) {
      v16bf pa;
      const int kA = kb + ((lane >> 4) << 3);
#pragma unroll
      for (int e = 0; e < 8; ++e) {
        pa[e]     = (bf16)(__expf(srow[kA + e] - mx) * inv);
        pa[e + 8] = (bf16)(__expf(srow[kA + 16 + e] - mx) * inv);
      }
      v16bf vb = load_fragB(vbase + kb, N_, lane);
      acc = wmma_bf16(pa, vb, acc);
    }
#pragma unroll
    for (int r = 0; r < 8; ++r) {
      const int m = r + ((lane >> 4) << 3);
      const int nq = q0 + m;
      o[(size_t)(bb * N_ + nq) * D_ + hh * HD_ + nt * 16 + (lane & 15)] =
          (bf16)acc[r];
    }
  }
}

// ---------------------------------------------------------------- launcher

extern "C" void kernel_launch(void* const* d_in, const int* in_sizes, int n_in,
                              void* d_out, int out_size, void* d_ws, size_t ws_size,
                              hipStream_t stream) {
  const float* x      = (const float*)d_in[0];
  const int*   mask   = (const int*)d_in[1];
  const float* inter  = (const float*)d_in[2];
  const float* w_qkv  = (const float*)d_in[3];
  const float* b_qkv  = (const float*)d_in[4];
  const float* w_out  = (const float*)d_in[5];
  const float* b_out  = (const float*)d_in[6];
  const float* w1     = (const float*)d_in[7];
  const float* b1     = (const float*)d_in[8];
  const float* w2     = (const float*)d_in[9];
  const float* b2     = (const float*)d_in[10];
  const float* g_pre  = (const float*)d_in[11];
  const float* be_pre = (const float*)d_in[12];
  const float* g_post = (const float*)d_in[13];
  const float* be_post= (const float*)d_in[14];
  const float* g_ffn  = (const float*)d_in[15];
  const float* be_ffn = (const float*)d_in[16];
  const float* g_mid  = (const float*)d_in[17];
  const float* be_mid = (const float*)d_in[18];
  float* out = (float*)d_out;

  char* ws = (char*)d_ws;
  const size_t MB = 1ull << 20;
  bf16* wtqkv = (bf16*)(ws + 0 * MB);     //  1.5 MB
  bf16* wtout = (bf16*)(ws + 2 * MB);     //  0.5 MB
  bf16* wt1   = (bf16*)(ws + 3 * MB);     //  2   MB
  bf16* wt2   = (bf16*)(ws + 6 * MB);     //  2   MB
  bf16* qkvb  = (bf16*)(ws + 8 * MB);     // 48  MB  (later: ffn1 64 MB, slot->80MB)
  bf16* ffn1b = (bf16*)(ws + 8 * MB);
  bf16* hbf   = (bf16*)(ws + 80 * MB);    // 16 MB   (later: f)
  bf16* fbf   = (bf16*)(ws + 80 * MB);
  bf16* vtb   = (bf16*)(ws + 96 * MB);    // 16 MB   (later: mid 64 MB -> 160MB)
  bf16* midb  = (bf16*)(ws + 96 * MB);
  bf16* obf   = (bf16*)(ws + 112 * MB);   // 16 MB
  float* oprj = (float*)(ws + 160 * MB);  // 32 MB
  float* attd = (float*)(ws + 192 * MB);  // 32 MB

  const int M = B_ * N_;                  // 16384

  // 1. weights -> bf16, transposed (Wt[n][k])
  convert_wt_kernel<<<(D_ * 3 * D_ + 255) / 256, 256, 0, stream>>>(w_qkv, wtqkv, D_, 3 * D_);
  convert_wt_kernel<<<(D_ * D_ + 255) / 256, 256, 0, stream>>>(w_out, wtout, D_, D_);
  convert_wt_kernel<<<(D_ * ED_ + 255) / 256, 256, 0, stream>>>(w1, wt1, D_, ED_);
  convert_wt_kernel<<<(ED_ * D_ + 255) / 256, 256, 0, stream>>>(w2, wt2, ED_, D_);

  // 2. h = LN(x; g_pre) -> bf16
  ln512_bf16_kernel<<<M, 128, 0, stream>>>(x, g_pre, be_pre, hbf);

  // 3. qkv = h @ w_qkv + b_qkv  (bf16 out)
  gemm_kernel<0><<<dim3(M / 128, (3 * D_) / 64), 256, 0, stream>>>(
      hbf, wtqkv, b_qkv, nullptr, qkvb, M, D_, 3 * D_);

  // 4. V transpose for attention B-fragments
  vt_kernel<<<(B_ * H_ * HD_ * N_) / 256, 256, 0, stream>>>(qkvb, vtb);

  // 5. attention
  attn_kernel<<<dim3(N_ / 64, H_, B_), 128, 0, stream>>>(qkvb, vtb, inter, mask, obf);

  // 6. o @ w_out + b_out -> f32
  gemm_kernel<1><<<dim3(M / 128, D_ / 64), 256, 0, stream>>>(
      obf, wtout, b_out, nullptr, oprj, M, D_, D_);

  // 7. attended = LN(oprj; g_post) + x ; f = LN(attended; g_ffn) bf16
  post_kernel<<<M, 128, 0, stream>>>(oprj, x, g_post, be_post, g_ffn, be_ffn,
                                     attd, fbf);

  // 8. ffn1 = gelu(f @ w1 + b1) -> bf16
  gemm_kernel<2><<<dim3(M / 128, ED_ / 64), 256, 0, stream>>>(
      fbf, wt1, b1, nullptr, ffn1b, M, D_, ED_);

  // 9. mid = LN(ffn1; g_mid) -> bf16
  ln2048_bf16_kernel<<<M, 256, 0, stream>>>(ffn1b, g_mid, be_mid, midb);

  // 10. out = mid @ w2 + b2 + attended
  gemm_kernel<3><<<dim3(M / 128, D_ / 64), 256, 0, stream>>>(
      midb, wt2, b2, attd, out, M, ED_, D_);
}
